// Lorsa_25615184953989
// MI455X (gfx1250) — matrix operand order repaired and checked by reading, not compiled, not run on hardware
//
#include <hip/hip_runtime.h>
#include <hip/hip_bf16.h>
#include <stdint.h>

// ---------- types ----------
typedef __bf16 bf16x16 __attribute__((ext_vector_type(16)));
typedef float  f32x8   __attribute__((ext_vector_type(8)));
typedef unsigned int u32x4 __attribute__((ext_vector_type(4)));
typedef int          i32x4 __attribute__((ext_vector_type(4)));
typedef int          i32x8 __attribute__((ext_vector_type(8)));

union FragAB { bf16x16 v; uint4 q[2]; };

#if defined(__HIP_DEVICE_COMPILE__) && __has_builtin(__builtin_amdgcn_tensor_load_to_lds) && __has_builtin(__builtin_amdgcn_s_wait_tensorcnt)
#define USE_TDM 1
#else
#define USE_TDM 0
#endif

__device__ __forceinline__ unsigned short f2bf(float f) {
    unsigned int u = __float_as_uint(f);
    u += 0x7FFFu + ((u >> 16) & 1u);   // round-to-nearest-even
    return (unsigned short)(u >> 16);
}
__device__ __forceinline__ float bf2f(unsigned short h) {
    return __uint_as_float(((unsigned int)h) << 16);
}

#if USE_TDM
// Issue one TDM 2D tile load: tile_dim0=32 elems (bf16), tile_dim1=128 rows,
// row-major into LDS at lds_byte_off. Tracked on TENSORcnt.
__device__ __forceinline__ void tdm_load_tile_128x32(unsigned lds_byte_off,
                                                     const unsigned short* gtile,
                                                     unsigned tensor_rows,
                                                     unsigned row_len_elems) {
    unsigned long long ga = (unsigned long long)(uintptr_t)gtile;
    u32x4 g0;
    g0.x = 1u;                                            // count=1, no gather
    g0.y = lds_byte_off;                                  // lds_addr (bytes)
    g0.z = (unsigned)(ga & 0xFFFFFFFFu);                  // global_addr[31:0]
    g0.w = (unsigned)((ga >> 32) & 0x1FFFFFFu) | (2u << 30); // addr[56:32] | type=2
    i32x8 g1;
    g1[0] = 0x00010000;                                   // data_size=1 (2 bytes)
    g1[1] = (int)((row_len_elems & 0xFFFFu) << 16);       // tensor_dim0[15:0]
    g1[2] = (int)(((row_len_elems >> 16) & 0xFFFFu) |     // tensor_dim0[31:16]
                  ((tensor_rows & 0xFFFFu) << 16));       // tensor_dim1[15:0]
    g1[3] = (int)(((tensor_rows >> 16) & 0xFFFFu) |       // tensor_dim1[31:16]
                  (32u << 16));                           // tile_dim0 = 32
    g1[4] = 128;                                          // tile_dim1 = 128, tile_dim2 = 0
    g1[5] = (int)row_len_elems;                           // tensor_dim0_stride[31:0]
    g1[6] = 0;                                            // stride hi / dim1_stride lo
    g1[7] = 0;
    i32x4 gz4; gz4[0] = 0; gz4[1] = 0; gz4[2] = 0; gz4[3] = 0;   // groups 2/3 unused (2D)
    i32x8 gz8;
#pragma unroll
    for (int i = 0; i < 8; ++i) gz8[i] = 0;
    __builtin_amdgcn_tensor_load_to_lds(g0, g1, gz4, gz4, gz8, 0);
}
#endif

// ---------- fp32 -> bf16 conversion ----------
__global__ void cvt_f32_bf16(const float* __restrict__ in, unsigned short* __restrict__ out, int n) {
    int i = blockIdx.x * blockDim.x + threadIdx.x;
    if (i < n) out[i] = f2bf(in[i]);
}

// ---------- mean of sparse weights ----------
__global__ void meanw_kernel(const float* __restrict__ sWV, const float* __restrict__ sWO,
                             float* __restrict__ vw, float* __restrict__ ow) {
    int i = blockIdx.x * blockDim.x + threadIdx.x;
    if (i >= 16 * 64) return;
    int h = i >> 6, d = i & 63;
    float s = 0.f;
    for (int k = 0; k < 128; ++k) s += sWV[((size_t)h * 64 + d) * 128 + k];
    vw[i] = s * (1.f / 128.f);
    float t = 0.f;
    for (int k = 0; k < 128; ++k) t += sWO[((size_t)h * 128 + k) * 64 + d];
    ow[i] = t * (1.f / 128.f);
}

// ---------- bf16 WMMA GEMM: C[M,N] = A[M,K] * W[N,K]^T ----------
// mode 0: f32 row-major out.  mode 1: bf16 out in [B,nh,S,hd].  mode 2: mode1 + per-head LayerNorm.
// Staging: TDM (tensor_load_to_lds) double-buffered when available, else manual copy.
#define GEMM_BM 128
#define GEMM_BN 128
#define GEMM_BK 32
#define GEMM_LDS_BYTES (4 * GEMM_BM * GEMM_BK * 2)   // 2 buffers x (A+B) = 32 KB

__global__ __launch_bounds__(256)
void gemm_bf16_wmma(const unsigned short* __restrict__ A,
                    const unsigned short* __restrict__ W,
                    float* __restrict__ outF,
                    unsigned short* __restrict__ outH,
                    const float* __restrict__ lnw, const float* __restrict__ lnb,
                    int M, int N, int K, int mode, float oscale, int S) {
    extern __shared__ __align__(16) char smem[];
    unsigned short* As0 = (unsigned short*)(smem);
    unsigned short* Bs0 = (unsigned short*)(smem + 8192);
    unsigned short* As1 = (unsigned short*)(smem + 16384);
    unsigned short* Bs1 = (unsigned short*)(smem + 24576);

    const int tid  = threadIdx.x;
    const int wave = tid >> 5;
    const int lane = tid & 31;
    const int half = lane >> 4;
    const int l    = lane & 15;
    const int wm = (wave >> 1) * 32;   // wave M offset in tile: 0,32,64,96
    const int wn = (wave & 1) * 64;    // wave N offset in tile: 0,64 (== one head)
    const int m0 = blockIdx.y * GEMM_BM;
    const int n0 = blockIdx.x * GEMM_BN;

#if USE_TDM
    const unsigned ldsbase = (unsigned)__builtin_amdgcn_groupstaticsize();
#endif

    f32x8 acc[2][4];
#pragma unroll
    for (int i = 0; i < 2; ++i)
#pragma unroll
        for (int j = 0; j < 4; ++j) acc[i][j] = {};

#if USE_TDM
    auto stage = [&](int k0s, int buf) {
        if (wave == 0) {
            tdm_load_tile_128x32(ldsbase + (buf ? 16384u : 0u),
                                 &A[(size_t)m0 * K + k0s], (unsigned)M, (unsigned)K);
            tdm_load_tile_128x32(ldsbase + (buf ? 24576u : 8192u),
                                 &W[(size_t)n0 * K + k0s], (unsigned)N, (unsigned)K);
        }
    };
#else
    auto stage = [&](int k0s, int buf) {
        unsigned short* Ad = buf ? As1 : As0;
        unsigned short* Bd = buf ? Bs1 : Bs0;
#pragma unroll
        for (int i = 0; i < 2; ++i) {
            int idx = tid + i * 256;     // 512 chunks of 8 bf16
            int row = idx >> 2;
            int c8  = (idx & 3) * 8;
            *(uint4*)&Ad[row * GEMM_BK + c8] = *(const uint4*)&A[(size_t)(m0 + row) * K + k0s + c8];
            *(uint4*)&Bd[row * GEMM_BK + c8] = *(const uint4*)&W[(size_t)(n0 + row) * K + k0s + c8];
        }
    };
#endif

    stage(0, 0);
    int cur = 0;
    for (int k0 = 0; k0 < K; k0 += GEMM_BK) {
        const bool more = (k0 + GEMM_BK) < K;
        if (more) stage(k0 + GEMM_BK, cur ^ 1);  // prefetch next stage into other buffer
#if USE_TDM
        if (wave == 0) {
            if (more) __builtin_amdgcn_s_wait_tensorcnt(2);  // current stage done, next in flight
            else      __builtin_amdgcn_s_wait_tensorcnt(0);
        }
#endif
        __syncthreads();
        const unsigned short* Asb = cur ? As1 : As0;
        const unsigned short* Bsb = cur ? Bs1 : Bs0;

        FragAB a[2], b[4];
#pragma unroll
        for (int mt = 0; mt < 2; ++mt) {
            // A frag: half0 -> k 0..7 & 16..23, half1 -> k 8..15 & 24..31
            const unsigned short* p = &Asb[(wm + mt * 16 + l) * GEMM_BK + half * 8];
            a[mt].q[0] = *(const uint4*)p;
            a[mt].q[1] = *(const uint4*)(p + 16);
        }
#pragma unroll
        for (int nt = 0; nt < 4; ++nt) {
            // B frag: lane = column, half0 -> k 0..7 & 16..23, half1 -> k 8..15 & 24..31
            const unsigned short* p = &Bsb[(wn + nt * 16 + l) * GEMM_BK + half * 8];
            b[nt].q[0] = *(const uint4*)p;
            b[nt].q[1] = *(const uint4*)(p + 16);
        }
#pragma unroll
        for (int mt = 0; mt < 2; ++mt)
#pragma unroll
            for (int nt = 0; nt < 4; ++nt)
                acc[mt][nt] = __builtin_amdgcn_wmma_f32_16x16x32_bf16(
                    false, a[mt].v, false, b[nt].v, (short)0, acc[mt][nt], false, false);
        cur ^= 1;
        __syncthreads();   // all reads done before this buffer is restaged
    }

    const int nh = 16, hd = 64;
    if (mode == 0) {
#pragma unroll
        for (int mt = 0; mt < 2; ++mt)
#pragma unroll
            for (int nt = 0; nt < 4; ++nt)
#pragma unroll
                for (int v = 0; v < 8; ++v) {
                    int m = m0 + wm + mt * 16 + half * 8 + v;
                    int n = n0 + wn + nt * 16 + l;
                    outF[(size_t)m * N + n] = acc[mt][nt][v] * oscale;
                }
    } else {
        const int h = (n0 + wn) >> 6;   // wave's 64 columns == one head exactly
#pragma unroll
        for (int mt = 0; mt < 2; ++mt) {
            float mu[8], rs[8];
#pragma unroll
            for (int v = 0; v < 8; ++v) { mu[v] = 0.f; rs[v] = 1.f; }
            if (mode == 2) {
                float sum[8], sq[8];
#pragma unroll
                for (int v = 0; v < 8; ++v) {
                    float s = 0.f, q = 0.f;
#pragma unroll
                    for (int nt = 0; nt < 4; ++nt) { float x = acc[mt][nt][v]; s += x; q += x * x; }
                    sum[v] = s; sq[v] = q;
                }
#pragma unroll
                for (int mask = 1; mask <= 8; mask <<= 1)
#pragma unroll
                    for (int v = 0; v < 8; ++v) {
                        sum[v] += __shfl_xor(sum[v], mask, 32);
                        sq[v]  += __shfl_xor(sq[v],  mask, 32);
                    }
#pragma unroll
                for (int v = 0; v < 8; ++v) {
                    float m_ = sum[v] * (1.f / 64.f);
                    float var = sq[v] * (1.f / 64.f) - m_ * m_;
                    mu[v] = m_;
                    rs[v] = rsqrtf(var + 1e-5f);
                }
            }
#pragma unroll
            for (int nt = 0; nt < 4; ++nt) {
                int d = nt * 16 + l;
                float w = (mode == 2) ? lnw[d] : 1.f;
                float bb = (mode == 2) ? lnb[d] : 0.f;
#pragma unroll
                for (int v = 0; v < 8; ++v) {
                    int m = m0 + wm + mt * 16 + half * 8 + v;
                    int bidx = m / S, s = m % S;
                    float x = acc[mt][nt][v];
                    if (mode == 2) x = (x - mu[v]) * rs[v] * w + bb;
                    x *= oscale;
                    outH[(((size_t)bidx * nh + h) * S + s) * hd + d] = f2bf(x);
                }
            }
        }
    }
}

// ---------- flash attention: 64-query block, 64-key blocks, bf16 WMMA ----------
#define AQ  64
#define AKB 64

__global__ __launch_bounds__(256)
void attn_wmma(const unsigned short* __restrict__ Qb,
               const unsigned short* __restrict__ Kb,
               const unsigned short* __restrict__ Vb,
               unsigned short* __restrict__ ctx,
               int S) {
    __shared__ unsigned short Qs[AQ * 64];       // 8 KB
    __shared__ unsigned short Ks[AKB * 64];      // 8 KB
    __shared__ unsigned short Vt[64 * AKB];      // 8 KB, transposed [d][key]
    __shared__ float          Sl[AQ * AKB];      // 16 KB
    __shared__ unsigned short Pl[AQ * AKB];      // 8 KB
    __shared__ float mrow[AQ], lrow[AQ], srow[AQ];

    const int tid = threadIdx.x;
    const int wave = tid >> 5, lane = tid & 31;
    const int half = lane >> 4, l = lane & 15;
    const int mtile = wave >> 1;                 // 0..3 -> 16 query rows each
    const int nhalf = wave & 1;                  // 0..1 -> 32 cols each
    const int qblk = blockIdx.x, h = blockIdx.y, b = blockIdx.z;
    const int nh = 16, hd = 64;
    const size_t headBase = ((size_t)b * nh + h) * S;
    const int q0 = qblk * AQ;

    // stage Q block (rows hold Q already scaled by 1/sqrt(hd) from projection)
#pragma unroll
    for (int i = 0; i < 2; ++i) {
        int idx = tid + i * 256;
        int r = idx >> 3;
        int c = (idx & 7) * 8;
        *(uint4*)&Qs[r * 64 + c] = *(const uint4*)&Qb[(headBase + q0 + r) * 64 + c];
    }
    if (tid < AQ) { mrow[tid] = -1e30f; lrow[tid] = 0.f; }

    f32x8 acc[2];
    acc[0] = {}; acc[1] = {};

    for (int kb = 0; kb < S; kb += AKB) {
        __syncthreads();
        // stage K (row-major) and V (transposed into Vt[d][key])
#pragma unroll
        for (int i = 0; i < 2; ++i) {
            int idx = tid + i * 256;
            int r = idx >> 3;               // key row
            int c = (idx & 7) * 8;          // d offset
            *(uint4*)&Ks[r * 64 + c] = *(const uint4*)&Kb[(headBase + kb + r) * 64 + c];
            uint4 vv = *(const uint4*)&Vb[(headBase + kb + r) * 64 + c];
            const unsigned short* ve = (const unsigned short*)&vv;
#pragma unroll
            for (int j = 0; j < 8; ++j) Vt[(c + j) * AKB + r] = ve[j];
        }
        __syncthreads();

        // scores: wave region = rows [mtile*16,+16) x keys [nhalf*32,+32)
        f32x8 st[2];
        st[0] = {}; st[1] = {};
#pragma unroll
        for (int ks = 0; ks < 2; ++ks) {
            FragAB aq;
            const unsigned short* pa = &Qs[(mtile * 16 + l) * 64 + ks * 32 + half * 8];
            aq.q[0] = *(const uint4*)pa;
            aq.q[1] = *(const uint4*)(pa + 16);
#pragma unroll
            for (int nt = 0; nt < 2; ++nt) {
                int col = nhalf * 32 + nt * 16 + l;    // key index in block
                FragAB bk;
                const unsigned short* pb = &Ks[col * 64 + ks * 32 + half * 8];
                bk.q[0] = *(const uint4*)pb;
                bk.q[1] = *(const uint4*)(pb + 16);
                st[nt] = __builtin_amdgcn_wmma_f32_16x16x32_bf16(
                    false, aq.v, false, bk.v, (short)0, st[nt], false, false);
            }
        }
#pragma unroll
        for (int nt = 0; nt < 2; ++nt)
#pragma unroll
            for (int v = 0; v < 8; ++v)
                Sl[(mtile * 16 + half * 8 + v) * AKB + nhalf * 32 + nt * 16 + l] = st[nt][v];
        __syncthreads();

        // online softmax (one thread per query row)
        if (tid < AQ) {
            float mold = mrow[tid];
            float bm = -1e30f;
            for (int j = 0; j < AKB; ++j) bm = fmaxf(bm, Sl[tid * AKB + j]);
            float mnew = fmaxf(mold, bm);
            float corr = __expf(mold - mnew);
            float psum = 0.f;
            for (int j = 0; j < AKB; ++j) {
                float p = __expf(Sl[tid * AKB + j] - mnew);
                Pl[tid * AKB + j] = f2bf(p);
                psum += p;
            }
            mrow[tid] = mnew;
            lrow[tid] = lrow[tid] * corr + psum;
            srow[tid] = corr;
        }
        __syncthreads();

        // rescale running context, then context += P @ V
#pragma unroll
        for (int v = 0; v < 8; ++v) {
            float sc = srow[mtile * 16 + half * 8 + v];
            acc[0][v] *= sc;
            acc[1][v] *= sc;
        }
#pragma unroll
        for (int ks = 0; ks < 2; ++ks) {
            FragAB ap;
            const unsigned short* pa = &Pl[(mtile * 16 + l) * AKB + ks * 32 + half * 8];
            ap.q[0] = *(const uint4*)pa;
            ap.q[1] = *(const uint4*)(pa + 16);
#pragma unroll
            for (int nt = 0; nt < 2; ++nt) {
                int col = nhalf * 32 + nt * 16 + l;    // d index
                FragAB bv;
                const unsigned short* pb = &Vt[col * AKB + ks * 32 + half * 8];
                bv.q[0] = *(const uint4*)pb;
                bv.q[1] = *(const uint4*)(pb + 16);
                acc[nt] = __builtin_amdgcn_wmma_f32_16x16x32_bf16(
                    false, ap.v, false, bv.v, (short)0, acc[nt], false, false);
            }
        }
    }
    __syncthreads();

    // write context in [B, S, nh, hd] (== context_flat [B*nh, S, hd] flat order)
#pragma unroll
    for (int v = 0; v < 8; ++v) {
        int r = mtile * 16 + half * 8 + v;
        float inv = 1.f / lrow[r];
        int s = q0 + r;
#pragma unroll
        for (int nt = 0; nt < 2; ++nt) {
            int d = nhalf * 32 + nt * 16 + l;
            ctx[(((size_t)b * S + s) * nh + h) * hd + d] = f2bf(acc[nt][v] * inv);
        }
    }
}

// ---------- z = ctx_flat[32,2048,64] . vw[16,64] ----------
__global__ void z_kernel(const unsigned short* __restrict__ ctx, const float* __restrict__ vw,
                         float* __restrict__ z) {
    int idx = blockIdx.x * blockDim.x + threadIdx.x;
    if (idx >= 32 * 16 * 2048) return;
    int n  = idx & 2047;
    int h  = (idx >> 11) & 15;
    int rr = idx >> 15;
    const unsigned short* c = &ctx[((size_t)rr * 2048 + n) * 64];
    const float* w = &vw[h * 64];
    float s = 0.f;
#pragma unroll
    for (int d = 0; d < 64; ++d) s += bf2f(c[d]) * w[d];
    z[idx] = s;
}

// ---------- top-128 along n (2048) per row via bisection threshold ----------
__global__ __launch_bounds__(256)
void topk_kernel(const float* __restrict__ z, float* __restrict__ zs) {
    __shared__ float row[2048];
    __shared__ float red[256];
    __shared__ int cnt;
    const int r = blockIdx.x;
    const int tid = threadIdx.x;
    const float* src = &z[(size_t)r * 2048];
    float lmax = -1e30f, lmin = 1e30f;
    for (int i = tid; i < 2048; i += 256) {
        float v = src[i]; row[i] = v;
        lmax = fmaxf(lmax, v); lmin = fminf(lmin, v);
    }
    red[tid] = lmax; __syncthreads();
    for (int s = 128; s > 0; s >>= 1) { if (tid < s) red[tid] = fmaxf(red[tid], red[tid + s]); __syncthreads(); }
    float hi = red[0]; __syncthreads();
    red[tid] = lmin; __syncthreads();
    for (int s = 128; s > 0; s >>= 1) { if (tid < s) red[tid] = fminf(red[tid], red[tid + s]); __syncthreads(); }
    float lo = red[0]; __syncthreads();

    for (int it = 0; it < 30; ++it) {
        float mid = 0.5f * (lo + hi);
        if (tid == 0) cnt = 0;
        __syncthreads();
        int c = 0;
        for (int i = tid; i < 2048; i += 256) c += (row[i] > mid) ? 1 : 0;
        atomicAdd(&cnt, c);
        __syncthreads();
        int total = cnt;
        __syncthreads();
        if (total >= 128) lo = mid; else hi = mid;
    }
    float thr = lo;   // count(> thr) converges to 128
    for (int i = tid; i < 2048; i += 256) {
        float v = row[i];
        zs[(size_t)r * 2048 + i] = (v > thr) ? v : 0.f;
    }
}

// ---------- y_flat = z_sparse . ow, with reshape/transpose into [B,S,H] bf16 ----------
__global__ void y_kernel(const float* __restrict__ zs, const float* __restrict__ ow,
                         unsigned short* __restrict__ ybf) {
    int idx = blockIdx.x * blockDim.x + threadIdx.x;
    if (idx >= 32 * 2048 * 64) return;
    int d  = idx & 63;
    int n  = (idx >> 6) & 2047;
    int rr = idx >> 17;        // y_flat row 0..31
    float s = 0.f;
#pragma unroll
    for (int h = 0; h < 16; ++h)
        s += zs[((size_t)rr * 16 + h) * 2048 + n] * ow[h * 64 + d];
    // flat idx == ((b2*16+h2)*2048+s2)*64+d of y_flat.reshape(B,nh,S,hd)
    int b2 = idx >> 21;
    int h2 = (idx >> 17) & 15;
    int s2 = (idx >> 6) & 2047;
    ybf[((size_t)b2 * 2048 + s2) * 1024 + h2 * 64 + d] = f2bf(s);
}

// ---------- host launcher ----------
extern "C" void kernel_launch(void* const* d_in, const int* in_sizes, int n_in,
                              void* d_out, int out_size, void* d_ws, size_t ws_size,
                              hipStream_t stream) {
    (void)in_sizes; (void)n_in; (void)out_size; (void)ws_size;
    const float* x   = (const float*)d_in[0];
    const float* WQ  = (const float*)d_in[1];
    const float* WK  = (const float*)d_in[2];
    const float* WV  = (const float*)d_in[3];
    const float* WO  = (const float*)d_in[4];
    const float* qlw = (const float*)d_in[5];
    const float* qlb = (const float*)d_in[6];
    const float* klw = (const float*)d_in[7];
    const float* klb = (const float*)d_in[8];
    const float* sWV = (const float*)d_in[9];
    const float* sWO = (const float*)d_in[10];

    const int B = 2, S = 2048, H = 1024, nh = 16;
    const int M = B * S, K = H, N = H;

    char* ws = (char*)d_ws;
    size_t off = 0;
    auto carve = [&](size_t bytes) -> char* {
        char* p = ws + off;
        off += (bytes + 255) & ~(size_t)255;
        return p;
    };
    unsigned short* xb  = (unsigned short*)carve((size_t)M * H * 2);
    unsigned short* Wqb = (unsigned short*)carve((size_t)H * H * 2);
    unsigned short* Wkb = (unsigned short*)carve((size_t)H * H * 2);
    unsigned short* Wvb = (unsigned short*)carve((size_t)H * H * 2);
    unsigned short* Wob = (unsigned short*)carve((size_t)H * H * 2);
    unsigned short* Qh  = (unsigned short*)carve((size_t)M * H * 2);
    unsigned short* Kh  = (unsigned short*)carve((size_t)M * H * 2);
    unsigned short* Vh  = (unsigned short*)carve((size_t)M * H * 2);
    unsigned short* ctx = (unsigned short*)carve((size_t)M * H * 2);
    float* vw = (float*)carve((size_t)nh * 64 * 4);
    float* ow = (float*)carve((size_t)nh * 64 * 4);
    float* z  = (float*)carve((size_t)32 * 16 * 2048 * 4);
    float* zs = (float*)carve((size_t)32 * 16 * 2048 * 4);
    unsigned short* ybf = (unsigned short*)carve((size_t)M * H * 2);

    // 1) bf16 conversions
    cvt_f32_bf16<<<(M * H + 255) / 256, 256, 0, stream>>>(x, xb, M * H);
    cvt_f32_bf16<<<(H * H + 255) / 256, 256, 0, stream>>>(WQ, Wqb, H * H);
    cvt_f32_bf16<<<(H * H + 255) / 256, 256, 0, stream>>>(WK, Wkb, H * H);
    cvt_f32_bf16<<<(H * H + 255) / 256, 256, 0, stream>>>(WV, Wvb, H * H);
    cvt_f32_bf16<<<(H * H + 255) / 256, 256, 0, stream>>>(WO, Wob, H * H);
    meanw_kernel<<<4, 256, 0, stream>>>(sWV, sWO, vw, ow);

    // 2) projections (LN fused into Q/K epilogues; 1/sqrt(hd) folded into Q)
    dim3 ggrid(N / GEMM_BN, M / GEMM_BM);
    gemm_bf16_wmma<<<ggrid, 256, GEMM_LDS_BYTES, stream>>>(xb, Wqb, nullptr, Qh, qlw, qlb, M, N, K, 2, 0.125f, S);
    gemm_bf16_wmma<<<ggrid, 256, GEMM_LDS_BYTES, stream>>>(xb, Wkb, nullptr, Kh, klw, klb, M, N, K, 2, 1.0f, S);
    gemm_bf16_wmma<<<ggrid, 256, GEMM_LDS_BYTES, stream>>>(xb, Wvb, nullptr, Vh, nullptr, nullptr, M, N, K, 1, 1.0f, S);

    // 3) attention
    dim3 agrid(S / AQ, nh, B);
    attn_wmma<<<agrid, 256, 0, stream>>>(Qh, Kh, Vh, ctx, S);

    // 4) sparse mixing path
    z_kernel<<<(32 * 16 * 2048) / 256, 256, 0, stream>>>(ctx, vw, z);
    topk_kernel<<<512, 256, 0, stream>>>(z, zs);
    y_kernel<<<(32 * 2048 * 64) / 256, 256, 0, stream>>>(zs, ow, ybf);

    // 5) output projection -> f32 d_out
    gemm_bf16_wmma<<<ggrid, 256, GEMM_LDS_BYTES, stream>>>(ybf, Wob, (float*)d_out, nullptr, nullptr, nullptr, M, N, K, 0, 1.0f, S);
}
